// CasFTSeqModel_42339787604239
// MI455X (gfx1250) — matrix operand order repaired and checked by reading, not compiled, not run on hardware
//
#include <hip/hip_runtime.h>
#include <hip/hip_bf16.h>

// ---------------- model dims ----------------
#define B_SZ   16384
#define BERT_D 768
#define ST     50      // sampling steps
#define TSTEPS 65      // GRU steps (TMAX+1)

typedef _Float16 fp16;
typedef __attribute__((ext_vector_type(16))) _Float16 v16h;
typedef __attribute__((ext_vector_type(8)))  float    v8f;

// ---------------- workspace layout (bytes) ----------------
#define OFF_SCHED   0u          // c1[50] @ +0, c2[50] @ +256, sv[50] @ +512 (floats)
#define OFF_TE      1024u       // te f16: 50*64
#define OFF_BPFRAG  8192u       // bp_w^T   frags K=768 N=64        -> 98304 B
#define OFF_W1FRAG  106496u     // dn_w1^T  frags K=160 N=256       -> 81920 B
#define OFF_W2FRAG  188416u     // dn_w2^T  frags K=256 N=256       -> 131072 B
#define OFF_W3FRAG  319488u     // dn_w3^T  frags K=256 N=16        -> 8192 B
#define OFF_WHFRAG  327680u     // [gru_wh|hd_w]^T frags K=128 N=416 -> 106496 B
#define OFF_COND16  434176u     // cond f16: 16384*64 -> 2 MB
#define OFF_H016    2531328u    // h0 f16: 16384*128  -> 4 MB

// ---------------- dynamic LDS sizes ----------------
#define COND_LDS 115968u
#define DDPM_LDS 287040u
#define GRU_LDS  242688u

// ---------------- WMMA helpers ----------------
__device__ __forceinline__ v8f wmma16(v16h a, v16h b, v8f c) {
  // D = A(16x32 f16) * B(32x16 f16) + C(16x16 f32)
  return __builtin_amdgcn_wmma_f32_16x16x32_f16(false, a, false, b, (short)0, c, false, false);
}

// Contiguous 32B fragment load (A or B operand stored in fragment order).
__device__ __forceinline__ v16h ldsFrag(const fp16* p) {
  return *(const v16h*)(p + (threadIdx.x & 31) * 16);
}

// Position (in halves) of element (row,k) inside an A-fragment-swizzled buffer
// with KT k-tiles per m-tile row. ISA 7.12.2 16-bit A 16x32 layout:
// lane = g*16 + (row&15), g = k[3]; element e = (k>>4)*8 + (k&7).
__device__ __forceinline__ int afrag_pos(int row, int k, int KT) {
  int mt = row >> 4, ml = row & 15;
  int kt = k >> 5, kl = k & 31;
  int g = (kl >> 3) & 1;
  int e = ((kl >> 4) << 3) | (kl & 7);
  return (((mt * KT) + kt) * 32 + (g << 4) + ml) * 16 + e;
}

__device__ __forceinline__ float silu_f(float x) { return x / (1.f + __expf(-x)); }
__device__ __forceinline__ float sigm_f(float x) { return 1.f / (1.f + __expf(-x)); }

// Row-major W (Nreal x Kin) -> WMMA B-fragment order for W^T (K x N), zero-padded.
__device__ void make_frags(const float* w, int Kin, int Nreal, int KT, int NT,
                           fp16* dst, int tid, int nthr) {
  int total = KT * NT * 512;
  for (int idx = tid; idx < total; idx += nthr) {
    int e    = idx & 15;
    int lane = (idx >> 4) & 31;
    int blk  = idx >> 9;
    int nt = blk % NT, kt = blk / NT;
    int k = kt * 32 + ((e >> 3) << 4) + ((lane >> 4) << 3) + (e & 7);
    int n = nt * 16 + (lane & 15);
    float v = (k < Kin && n < Nreal) ? w[n * Kin + k] : 0.f;
    dst[idx] = (fp16)v;
  }
}

__device__ __forceinline__ void cp16(const unsigned char* src, unsigned char* dst,
                                     int bytes, int tid, int nthr) {
  const uint4* s = (const uint4*)src;
  uint4* d = (uint4*)dst;
  for (int i = tid; i < bytes / 16; i += nthr) d[i] = s[i];
}

// ================= kernel 1: schedule + time MLP + weight fragging =================
__global__ void __launch_bounds__(256) prep_kernel(
    const float* __restrict__ tp_w1, const float* __restrict__ tp_b1,
    const float* __restrict__ tp_w2, const float* __restrict__ tp_b2,
    const float* __restrict__ bp_w,  const float* __restrict__ dn_w1,
    const float* __restrict__ dn_w2, const float* __restrict__ dn_w3,
    const float* __restrict__ gru_wh, const float* __restrict__ hd_w,
    unsigned char* __restrict__ ws) {
  int tid = threadIdx.x;
  float* sched = (float*)(ws + OFF_SCHED);
  fp16*  te16  = (fp16*)(ws + OFF_TE);

  if (tid == 0) {  // cosine schedule + DDPM coefficients (serial cumprod)
    const float PIH = 1.57079632679f;
    float betas[200];
    float abar[200];
    float ab_prev;
    {
      float x0 = 0.008f / 1.008f;
      float c0 = __cosf(x0 * PIH);
      ab_prev = c0 * c0;
    }
    float prod = 1.f;
    for (int i = 0; i < 200; ++i) {
      float x = (float)(i + 1) / 200.f;
      float c = __cosf((x + 0.008f) / 1.008f * PIH);
      float ab = c * c;
      float beta = 1.f - ab / ab_prev;
      if (beta > 0.9999f) beta = 0.9999f;
      betas[i] = beta;
      prod *= (1.f - beta);
      abar[i] = prod;
      ab_prev = ab;
    }
    for (int s = 0; s < ST; ++s) {
      int t = 199 - 4 * s;
      float abt = abar[t];
      float abp = (t > 0) ? abar[t - 1] : 1.f;
      float beta = betas[t];
      float alpha = 1.f - beta;
      sched[s]       = __fsqrt_rn(abp) * beta / (1.f - abt);
      sched[64 + s]  = __fsqrt_rn(alpha) * (1.f - abp) / (1.f - abt);
      float var = (1.f - abp) / (1.f - abt) * beta;
      sched[128 + s] = (t > 0) ? __fsqrt_rn(var) : 0.f;
    }
  }

  if (tid < ST) {  // sinusoidal emb + tiny 2-layer MLP per step
    int t = 199 - 4 * tid;
    float temb[64];
#pragma unroll
    for (int i = 0; i < 32; ++i) {
      float freq = __expf(-9.21034037198f * (float)i / 31.f);  // ln(10000)
      float e = (float)t * freq;
      temb[i] = __sinf(e);
      temb[32 + i] = __cosf(e);
    }
    float t1[128];
    for (int o = 0; o < 128; ++o) {
      float acc = tp_b1[o];
      for (int i = 0; i < 64; ++i) acc += tp_w1[o * 64 + i] * temb[i];
      t1[o] = silu_f(acc);
    }
    for (int o = 0; o < 64; ++o) {
      float acc = tp_b2[o];
      for (int i = 0; i < 128; ++i) acc += tp_w2[o * 128 + i] * t1[i];
      te16[tid * 64 + o] = (fp16)acc;
    }
  }

  make_frags(bp_w,  BERT_D, 64,  24, 4,  (fp16*)(ws + OFF_BPFRAG), tid, 256);
  make_frags(dn_w1, 138,    256, 5,  16, (fp16*)(ws + OFF_W1FRAG), tid, 256);
  make_frags(dn_w2, 256,    256, 8,  16, (fp16*)(ws + OFF_W2FRAG), tid, 256);
  make_frags(dn_w3, 256,    10,  8,  1,  (fp16*)(ws + OFF_W3FRAG), tid, 256);

  // combined recurrent + head weight: [gru_wh (384) | hd_w (2) | pad] -> N=416, K=128
  {
    fp16* dst = (fp16*)(ws + OFF_WHFRAG);
    int total = 4 * 26 * 512;
    for (int idx = tid; idx < total; idx += 256) {
      int e    = idx & 15;
      int lane = (idx >> 4) & 31;
      int blk  = idx >> 9;
      int nt = blk % 26, kt = blk / 26;
      int k = kt * 32 + ((e >> 3) << 4) + ((lane >> 4) << 3) + (e & 7);
      int n = nt * 16 + (lane & 15);
      float v = 0.f;
      if (n < 384) v = gru_wh[n * 128 + k];
      else if (n < 386) v = hd_w[(n - 384) * 128 + k];
      dst[idx] = (fp16)v;
    }
  }
}

// ================= kernel 2: cond = LayerNorm(bert @ bp_w^T + b) =================
__global__ void __launch_bounds__(256) cond_kernel(
    const float* __restrict__ bert, const float* __restrict__ bp_b,
    const float* __restrict__ ln_g, const float* __restrict__ ln_b,
    unsigned char* __restrict__ ws) {
  extern __shared__ unsigned char smem[];
  fp16*  bpf  = (fp16*)smem;                       // 98304 B
  float* cbuf = (float*)(smem + 98304);            // 64x64 f32
  float* mu   = (float*)(smem + 98304 + 16384);    // 64
  float* rs   = (float*)(smem + 98304 + 16384 + 256);
  float* prm  = (float*)(smem + 98304 + 16384 + 512);  // bp_b|ln_g|ln_b (3x64)

  int tid = threadIdx.x;
  cp16(ws + OFF_BPFRAG, (unsigned char*)bpf, 98304, tid, 256);
  if (tid < 64) {
    prm[tid]       = bp_b[tid];
    prm[64 + tid]  = ln_g[tid];
    prm[128 + tid] = ln_b[tid];
  }
  __syncthreads();

  int wave = tid >> 5, lane = tid & 31, g = lane >> 4;
  int m = wave >> 1, n0 = (wave & 1) * 2;   // 4 m-tiles x 4 n-tiles, 2 jobs/wave
  int row0 = blockIdx.x * 64;
  const float* abase = bert + (size_t)(row0 + m * 16 + (lane & 15)) * BERT_D;

  v8f acc0 = {}, acc1 = {};
#pragma unroll 2
  for (int kt = 0; kt < 24; ++kt) {
    __builtin_prefetch(abase + (kt + 2) * 32, 0, 1);
    v16h a;
#pragma unroll
    for (int v = 0; v < 8; ++v) {
      int k = kt * 32 + ((v >> 2) << 4) + (g << 3) + ((v & 3) << 1);
      float2 f = *(const float2*)(abase + k);
      a[2 * v]     = (fp16)f.x;
      a[2 * v + 1] = (fp16)f.y;
    }
    acc0 = wmma16(a, ldsFrag(bpf + (size_t)(kt * 4 + n0) * 512), acc0);
    acc1 = wmma16(a, ldsFrag(bpf + (size_t)(kt * 4 + n0 + 1) * 512), acc1);
  }
#pragma unroll
  for (int r = 0; r < 8; ++r) {
    int rowl = m * 16 + r + g * 8;
    int c0 = n0 * 16 + (lane & 15), c1 = (n0 + 1) * 16 + (lane & 15);
    cbuf[rowl * 64 + c0] = acc0[r] + prm[c0];
    cbuf[rowl * 64 + c1] = acc1[r] + prm[c1];
  }
  __syncthreads();

  if (tid < 64) {
    float s = 0.f, ss = 0.f;
    for (int i = 0; i < 64; ++i) {
      float v = cbuf[tid * 64 + i];
      s += v; ss += v * v;
    }
    float mean = s * (1.f / 64.f);
    float var = ss * (1.f / 64.f) - mean * mean;
    mu[tid] = mean;
    rs[tid] = __frsqrt_rn(var + 1e-5f);
  }
  __syncthreads();

  fp16* cond16 = (fp16*)(ws + OFF_COND16);
  for (int i = tid; i < 4096; i += 256) {
    int rowl = i >> 6, col = i & 63;
    float v = (cbuf[rowl * 64 + col] - mu[rowl]) * rs[rowl] * prm[64 + col] + prm[128 + col];
    cond16[(size_t)(row0 + rowl) * 64 + col] = (fp16)v;
  }
}

// ================= kernel 3: persistent 50-step DDPM denoiser =================
// Activations (inp, h) live in LDS in A-fragment-swizzled order so every WMMA
// A-operand is one contiguous 32B LDS read.
__global__ void __launch_bounds__(256) ddpm_kernel(
    const float* __restrict__ init_noise, const float* __restrict__ step_noise,
    const float* __restrict__ dn_b1, const float* __restrict__ dn_b2,
    const float* __restrict__ dn_b3, unsigned char* __restrict__ ws,
    float* __restrict__ out) {
  extern __shared__ unsigned char smem[];
  fp16*  w1f  = (fp16*)smem;                       // 81920  (K=160,N=256)
  fp16*  w2f  = (fp16*)(smem + 81920);             // 131072 (K=256,N=256)
  fp16*  w3f  = (fp16*)(smem + 212992);            // 8192   (K=256,N=16)
  fp16*  inpA = (fp16*)(smem + 221184);            // 64x160 (A-swizzled, KT=5)
  fp16*  hA   = (fp16*)(smem + 241664);            // 64x256 (A-swizzled, KT=8)
  float* xbuf = (float*)(smem + 274432);           // 64x16
  float* b1   = (float*)(smem + 278528);           // 256
  float* b2   = (float*)(smem + 279552);           // 256
  float* b3   = (float*)(smem + 280576);           // 16
  fp16*  tel  = (fp16*)(smem + 280640);            // 50x64

  int tid = threadIdx.x;
  int row0 = blockIdx.x * 64;

  cp16(ws + OFF_W1FRAG, (unsigned char*)w1f, 81920, tid, 256);
  cp16(ws + OFF_W2FRAG, (unsigned char*)w2f, 131072, tid, 256);
  cp16(ws + OFF_W3FRAG, (unsigned char*)w3f, 8192, tid, 256);
  cp16(ws + OFF_TE,     (unsigned char*)tel, 6400, tid, 256);
  if (tid < 256) { b1[tid] = dn_b1[tid]; b2[tid] = dn_b2[tid]; }
  if (tid < 16)  { b3[tid] = (tid < 10) ? dn_b3[tid] : 0.f; }

  {  // zero the whole input tile (covers K padding), then scatter cond columns
    uint4* z = (uint4*)inpA;
    for (int i = tid; i < 20480 / 16; i += 256) z[i] = (uint4){0, 0, 0, 0};
  }
  __syncthreads();
  const fp16* cond16 = (const fp16*)(ws + OFF_COND16);
  for (int i = tid; i < 4096; i += 256) {       // cond -> input cols 10..73
    int r = i >> 6, c = i & 63;
    inpA[afrag_pos(r, 10 + c, 5)] = cond16[(size_t)(row0 + r) * 64 + c];
  }
  for (int i = tid; i < 640; i += 256) {        // x <- init_noise
    int r = i / 10, c = i % 10;
    xbuf[r * 16 + c] = init_noise[(size_t)(row0 + r) * 10 + c];
  }
  __syncthreads();

  const float* sched = (const float*)(ws + OFF_SCHED);
  int wave = tid >> 5, lane = tid & 31, g = lane >> 4, col16 = lane & 15;
  int mb = (wave >> 2) * 2;          // m-tile pair base: 0 or 2
  int nb4 = (wave & 3) * 4;          // n-tile base: 0,4,8,12 (2m x 4n jobs/wave)

#pragma unroll 1
  for (int s = 0; s < ST; ++s) {
    // refresh x + time-embedding columns of the input tile
    for (int i = tid; i < 640; i += 256) {
      int r = i / 10, c = i % 10;
      inpA[afrag_pos(r, c, 5)] = (fp16)xbuf[r * 16 + c];
    }
    for (int i = tid; i < 4096; i += 256) {
      int r = i >> 6, c = i & 63;
      inpA[afrag_pos(r, 74 + c, 5)] = tel[s * 64 + c];
    }
    __syncthreads();

    // ---- layer 1: (64x160) @ (160x256), silu ----
    {
      v8f acc[2][4];
#pragma unroll
      for (int i = 0; i < 2; ++i)
#pragma unroll
        for (int j = 0; j < 4; ++j) acc[i][j] = (v8f){};
#pragma unroll
      for (int kt = 0; kt < 5; ++kt) {
        v16h a0 = ldsFrag(inpA + (size_t)(mb * 5 + kt) * 512);
        v16h a1 = ldsFrag(inpA + (size_t)((mb + 1) * 5 + kt) * 512);
#pragma unroll
        for (int j = 0; j < 4; ++j) {
          v16h b = ldsFrag(w1f + (size_t)(kt * 16 + nb4 + j) * 512);
          acc[0][j] = wmma16(a0, b, acc[0][j]);
          acc[1][j] = wmma16(a1, b, acc[1][j]);
        }
      }
#pragma unroll
      for (int i = 0; i < 2; ++i)
#pragma unroll
        for (int j = 0; j < 4; ++j) {
          int col = (nb4 + j) * 16 + col16;
#pragma unroll
          for (int r = 0; r < 8; ++r) {
            int rowl = (mb + i) * 16 + r + g * 8;
            hA[afrag_pos(rowl, col, 8)] = (fp16)silu_f(acc[i][j][r] + b1[col]);
          }
        }
    }
    __syncthreads();

    // ---- layer 2: (64x256) @ (256x256), silu (regs across barrier) ----
    {
      v8f acc[2][4];
#pragma unroll
      for (int i = 0; i < 2; ++i)
#pragma unroll
        for (int j = 0; j < 4; ++j) acc[i][j] = (v8f){};
#pragma unroll
      for (int kt = 0; kt < 8; ++kt) {
        v16h a0 = ldsFrag(hA + (size_t)(mb * 8 + kt) * 512);
        v16h a1 = ldsFrag(hA + (size_t)((mb + 1) * 8 + kt) * 512);
#pragma unroll
        for (int j = 0; j < 4; ++j) {
          v16h b = ldsFrag(w2f + (size_t)(kt * 16 + nb4 + j) * 512);
          acc[0][j] = wmma16(a0, b, acc[0][j]);
          acc[1][j] = wmma16(a1, b, acc[1][j]);
        }
      }
      __syncthreads();   // everyone done reading h1
#pragma unroll
      for (int i = 0; i < 2; ++i)
#pragma unroll
        for (int j = 0; j < 4; ++j) {
          int col = (nb4 + j) * 16 + col16;
#pragma unroll
          for (int r = 0; r < 8; ++r) {
            int rowl = (mb + i) * 16 + r + g * 8;
            hA[afrag_pos(rowl, col, 8)] = (fp16)silu_f(acc[i][j][r] + b2[col]);
          }
        }
    }
    __syncthreads();

    // ---- layer 3: (64x256) @ (256x16) + ancestral update (waves 0..3) ----
    if (wave < 4) {
      int m3 = wave;
      v8f acc = {};
#pragma unroll
      for (int kt = 0; kt < 8; ++kt)
        acc = wmma16(ldsFrag(hA + (size_t)(m3 * 8 + kt) * 512),
                     ldsFrag(w3f + (size_t)kt * 512), acc);
      if (col16 < 10) {
        float c1 = sched[s], c2 = sched[64 + s], sv = sched[128 + s];
#pragma unroll
        for (int r = 0; r < 8; ++r) {
          int rowl = m3 * 16 + r + g * 8;
          float x0p = acc[r] + b3[col16];
          float xo  = xbuf[rowl * 16 + col16];
          float nz  = step_noise[((size_t)s * B_SZ + row0 + rowl) * 10 + col16];
          xbuf[rowl * 16 + col16] = c1 * x0p + c2 * xo + sv * nz;
        }
      }
    }
    __syncthreads();
  }

  for (int i = tid; i < 640; i += 256) {
    int r = i / 10, c = i % 10;
    out[(size_t)(row0 + r) * 10 + c] = xbuf[r * 16 + c];
  }
}

// ================= kernel 4: seq_emb + h0 (tiny, VALU) =================
__global__ void __launch_bounds__(256) seqh0_kernel(
    const float* __restrict__ x0g,
    const float* __restrict__ sf_w1, const float* __restrict__ sf_b1,
    const float* __restrict__ sf_w2, const float* __restrict__ sf_b2,
    const float* __restrict__ sf_w3, const float* __restrict__ sf_b3,
    const float* __restrict__ h0_w,  const float* __restrict__ h0_b,
    unsigned char* __restrict__ ws) {
  int r = blockIdx.x * 256 + threadIdx.x;
  const fp16* cond16 = (const fp16*)(ws + OFF_COND16);
  fp16* h016 = (fp16*)(ws + OFF_H016);

  float x0v[10];
#pragma unroll
  for (int i = 0; i < 10; ++i) x0v[i] = x0g[(size_t)r * 10 + i];
  float e1[20];
  for (int o = 0; o < 20; ++o) {
    float a = sf_b1[o];
    for (int i = 0; i < 10; ++i) a += sf_w1[o * 10 + i] * x0v[i];
    e1[o] = fmaxf(a, 0.f);
  }
  float e2[20];
  for (int o = 0; o < 20; ++o) {
    float a = sf_b2[o];
    for (int i = 0; i < 20; ++i) a += sf_w2[o * 20 + i] * e1[i];
    e2[o] = fmaxf(a, 0.f);
  }
  float se[10];
  for (int o = 0; o < 10; ++o) {
    float a = sf_b3[o];
    for (int i = 0; i < 20; ++i) a += sf_w3[o * 20 + i] * e2[i];
    se[o] = a;
  }
  float cv[64];
#pragma unroll
  for (int i = 0; i < 64; ++i) cv[i] = (float)cond16[(size_t)r * 64 + i];
  for (int o = 0; o < 128; ++o) {
    float a = h0_b[o];
    for (int i = 0; i < 64; ++i) a += h0_w[o * 74 + i] * cv[i];
    for (int i = 0; i < 10; ++i) a += h0_w[o * 74 + 64 + i] * se[i];
    h016[(size_t)r * 128 + o] = (fp16)tanhf(a);
  }
}

// ================= kernel 5: persistent 65-step GRU decoder =================
// Head (hd_w) is folded into the recurrent GEMM as N-columns 384..385: the
// step-(s+1) GEMM emits the step-s head outputs; one tail pass covers s=64.
__global__ void __launch_bounds__(256) gru_kernel(
    const float* __restrict__ times,
    const float* __restrict__ gru_wi, const float* __restrict__ gru_bi,
    const float* __restrict__ gru_bh, const float* __restrict__ hd_b,
    unsigned char* __restrict__ ws, float* __restrict__ out) {
  extern __shared__ unsigned char smem[];
  fp16*  whf = (fp16*)smem;                   // 106496 (K=128,N=416 frags)
  float* gh  = (float*)(smem + 106496);       // 64x384
  fp16*  hbA = (fp16*)(smem + 204800);        // 64x128 (A-swizzled, KT=4)
  float* tl  = (float*)(smem + 221184);       // 64x64 times
  float* wi  = (float*)(smem + 237568);       // 384
  float* bi  = (float*)(smem + 239104);       // 384
  float* bh  = (float*)(smem + 240640);       // 384
  float* hdb = (float*)(smem + 242176);       // 2

  int tid = threadIdx.x;
  int row0 = blockIdx.x * 64;

  cp16(ws + OFF_WHFRAG, (unsigned char*)whf, 106496, tid, 256);
  {
    const fp16* h016 = (const fp16*)(ws + OFF_H016) + (size_t)row0 * 128;
    for (int i = tid; i < 8192; i += 256) {
      int row = i >> 7, j = i & 127;
      hbA[afrag_pos(row, j, 4)] = h016[row * 128 + j];
    }
  }
  for (int i = tid; i < 4096; i += 256) tl[i] = times[(size_t)row0 * 64 + i];
  for (int i = tid; i < 384; i += 256) { wi[i] = gru_wi[i]; bi[i] = gru_bi[i]; bh[i] = gru_bh[i]; }
  if (tid < 2) hdb[tid] = hd_b[tid];
  __syncthreads();

  float* predg = out + (size_t)B_SZ * 10;
  float* stopg = predg + (size_t)B_SZ * TSTEPS;

  int wave = tid >> 5, lane = tid & 31, g = lane >> 4, col16 = lane & 15;
  int m = wave >> 1, nb = (wave & 1) * 13;    // 4m x 26n tiles, 13 jobs/wave

#pragma unroll 1
  for (int s = 0; s < TSTEPS; ++s) {
    // ---- [gh | head(s-1)] = h @ [wh | hd]^T : (64x128) @ (128x416) ----
    {
      v8f acc[13];
#pragma unroll
      for (int j = 0; j < 13; ++j) acc[j] = (v8f){};
#pragma unroll
      for (int kt = 0; kt < 4; ++kt) {
        v16h a = ldsFrag(hbA + (size_t)(m * 4 + kt) * 512);
#pragma unroll
        for (int j = 0; j < 13; ++j)
          acc[j] = wmma16(a, ldsFrag(whf + (size_t)(kt * 26 + nb + j) * 512), acc[j]);
      }
#pragma unroll
      for (int j = 0; j < 13; ++j) {
        int nt = nb + j;
        if (nt < 24) {
          int col = nt * 16 + col16;
#pragma unroll
          for (int r = 0; r < 8; ++r)
            gh[(m * 16 + r + g * 8) * 384 + col] = acc[j][r];
        } else if (nt == 24) {
          if (s > 0 && col16 < 2) {
            float* dst = (col16 == 0) ? predg : stopg;
#pragma unroll
            for (int r = 0; r < 8; ++r) {
              int row = m * 16 + r + g * 8;
              dst[(size_t)(row0 + row) * TSTEPS + (s - 1)] = acc[j][r] + hdb[col16];
            }
          }
        }  // nt == 25: zero padding, discard
      }
    }
    __syncthreads();

    // ---- gates + state update ----
    for (int i = tid; i < 8192; i += 256) {
      int row = i >> 7, j = i & 127;
      float xin = 0.f;
      if (s == 1) xin = tl[row * 64];
      else if (s >= 2) xin = tl[row * 64 + s - 1] - tl[row * 64 + s - 2];
      float gr = xin * wi[j]       + bi[j]       + gh[row * 384 + j]       + bh[j];
      float gz = xin * wi[128 + j] + bi[128 + j] + gh[row * 384 + 128 + j] + bh[128 + j];
      float gni = xin * wi[256 + j] + bi[256 + j];
      float ghn = gh[row * 384 + 256 + j] + bh[256 + j];
      float rg = sigm_f(gr);
      float zg = sigm_f(gz);
      float ng = tanhf(gni + rg * ghn);
      int hp = afrag_pos(row, j, 4);
      float hold = (float)hbA[hp];
      hbA[hp] = (fp16)((1.f - zg) * ng + zg * hold);
    }
    __syncthreads();
  }

  // tail: head outputs for the final step (s = 64) from h_64
  if (wave < 4) {
    int m3 = wave;
    v8f acc = {};
#pragma unroll
    for (int kt = 0; kt < 4; ++kt)
      acc = wmma16(ldsFrag(hbA + (size_t)(m3 * 4 + kt) * 512),
                   ldsFrag(whf + (size_t)(kt * 26 + 24) * 512), acc);
    if (col16 < 2) {
      float* dst = (col16 == 0) ? predg : stopg;
#pragma unroll
      for (int r = 0; r < 8; ++r) {
        int row = m3 * 16 + r + g * 8;
        dst[(size_t)(row0 + row) * TSTEPS + (TSTEPS - 1)] = acc[r] + hdb[col16];
      }
    }
  }
}

// ================= launch =================
extern "C" void kernel_launch(void* const* d_in, const int* in_sizes, int n_in,
                              void* d_out, int out_size, void* d_ws, size_t ws_size,
                              hipStream_t stream) {
  (void)in_sizes; (void)n_in; (void)out_size; (void)ws_size;
  const float* bert       = (const float*)d_in[0];
  const float* times      = (const float*)d_in[1];
  const float* init_noise = (const float*)d_in[2];
  const float* step_noise = (const float*)d_in[3];
  const float* bp_w  = (const float*)d_in[4];
  const float* bp_b  = (const float*)d_in[5];
  const float* ln_g  = (const float*)d_in[6];
  const float* ln_b  = (const float*)d_in[7];
  const float* tp_w1 = (const float*)d_in[8];
  const float* tp_b1 = (const float*)d_in[9];
  const float* tp_w2 = (const float*)d_in[10];
  const float* tp_b2 = (const float*)d_in[11];
  const float* dn_w1 = (const float*)d_in[12];
  const float* dn_b1 = (const float*)d_in[13];
  const float* dn_w2 = (const float*)d_in[14];
  const float* dn_b2 = (const float*)d_in[15];
  const float* dn_w3 = (const float*)d_in[16];
  const float* dn_b3 = (const float*)d_in[17];
  const float* sf_w1 = (const float*)d_in[18];
  const float* sf_b1 = (const float*)d_in[19];
  const float* sf_w2 = (const float*)d_in[20];
  const float* sf_b2 = (const float*)d_in[21];
  const float* sf_w3 = (const float*)d_in[22];
  const float* sf_b3 = (const float*)d_in[23];
  const float* h0_w  = (const float*)d_in[24];
  const float* h0_b  = (const float*)d_in[25];
  const float* gru_wi = (const float*)d_in[26];
  const float* gru_bi = (const float*)d_in[27];
  const float* gru_wh = (const float*)d_in[28];
  const float* gru_bh = (const float*)d_in[29];
  const float* hd_w  = (const float*)d_in[30];
  const float* hd_b  = (const float*)d_in[31];
  unsigned char* ws = (unsigned char*)d_ws;
  float* out = (float*)d_out;

  prep_kernel<<<1, 256, 0, stream>>>(tp_w1, tp_b1, tp_w2, tp_b2,
                                     bp_w, dn_w1, dn_w2, dn_w3, gru_wh, hd_w, ws);
  cond_kernel<<<B_SZ / 64, 256, COND_LDS, stream>>>(bert, bp_b, ln_g, ln_b, ws);
  ddpm_kernel<<<B_SZ / 64, 256, DDPM_LDS, stream>>>(init_noise, step_noise,
                                                    dn_b1, dn_b2, dn_b3, ws, out);
  seqh0_kernel<<<B_SZ / 256, 256, 0, stream>>>(out, sf_w1, sf_b1, sf_w2, sf_b2,
                                               sf_w3, sf_b3, h0_w, h0_b, ws);
  gru_kernel<<<B_SZ / 64, 256, GRU_LDS, stream>>>(times, gru_wi, gru_bi, gru_bh,
                                                  hd_b, ws, out);
}
